// SelfAttention_56693568307268
// MI455X (gfx1250) — compile-verified, matrix operand order checked by reading, not run on hardware
//
#include <hip/hip_runtime.h>
#include <stdint.h>

#define SEQ 8192
#define DM  768

typedef unsigned short u16;
typedef __attribute__((ext_vector_type(8)))  u16    u16x8;
typedef __attribute__((ext_vector_type(16))) u16    u16x16;
typedef __attribute__((ext_vector_type(16))) __bf16 bf16x16;
typedef __attribute__((ext_vector_type(8)))  float  f32x8;
typedef __attribute__((ext_vector_type(4)))  float  f32x4;
typedef __attribute__((ext_vector_type(4)))  int    i32x4;

#if defined(__gfx1250__) && __has_builtin(__builtin_amdgcn_global_load_async_to_lds_b128)
#define HAVE_ASYNC_LDS 1
typedef __attribute__((address_space(1))) i32x4 gas_i32x4;   // global
typedef __attribute__((address_space(3))) i32x4 las_i32x4;   // LDS
#endif

// Native bf16 convert -> v_cvt_*bf16_f32 on gfx1250 (replaces 3-4 VALU ops of
// manual round-to-nearest with one hardware convert).
static __device__ __forceinline__ u16 f32_to_bf16(float f) {
  return __builtin_bit_cast(u16, (__bf16)f);
}

// Per-lane bf16 fragment slice for v_wmma_f32_16x16x32_bf16.
// ISA layout: lane-half h holds K = 8h..8h+7 (elems 0..7) and 16+8h..23+8h (elems 8..15).
// row points at this lane's 32-element K slab (LDS or global), 16B aligned.
static __device__ __forceinline__ bf16x16 load_frag(const u16* __restrict__ row, int half) {
  u16x8 lo = *(const u16x8*)(row + 8 * half);
  u16x8 hi = *(const u16x8*)(row + 16 + 8 * half);
  u16x16 v = __builtin_shufflevector(lo, hi, 0,1,2,3,4,5,6,7,8,9,10,11,12,13,14,15);
  return __builtin_bit_cast(bf16x16, v);
}

static __device__ __forceinline__ f32x8 wmma_bf16(bf16x16 a, bf16x16 b, f32x8 c) {
  return __builtin_amdgcn_wmma_f32_16x16x32_bf16(
      /*neg_a=*/false, a, /*neg_b=*/false, b,
      /*c_mod=*/(short)0, c, /*reuse_a=*/false, /*reuse_b=*/false);
}

// ---------------------------------------------------------------------------
// Kernel 1: Q = x @ Wq^T, K = x @ Wk^T, V = x @ Wv^T  (f32 in -> bf16 out)
// Q,K stored row-major [SEQ][DM]; V stored transposed Vt [DM][SEQ] so the
// attention kernel's P@V B-fragments are two contiguous 16B loads per lane.
// Grid: (DM/64, SEQ/64, 3), block 256 (8 waves, each owns two 16x16 tiles).
// ---------------------------------------------------------------------------
__global__ __launch_bounds__(256) void qkv_proj_kernel(
    const float* __restrict__ x,  const float* __restrict__ wq,
    const float* __restrict__ wk, const float* __restrict__ wv,
    u16* __restrict__ Qb, u16* __restrict__ Kb, u16* __restrict__ Vt)
{
  __shared__ u16 Xs[64][32];
  __shared__ u16 Ws[64][32];

  const int tid  = threadIdx.x;
  const int lane = tid & 31;
  const int wid  = tid >> 5;
  const int half = lane >> 4;
  const int ln   = lane & 15;

  const int n0 = blockIdx.x * 64;
  const int m0 = blockIdx.y * 64;
  const int which = blockIdx.z;
  const float* __restrict__ W = (which == 0) ? wq : ((which == 1) ? wk : wv);

  const int mt  = wid & 3;          // 16-row tile within 64
  const int nt0 = (wid >> 2) * 2;   // first of two 16-col tiles

  f32x8 acc0 = {}, acc1 = {};

  const int lrow = tid >> 2;        // 0..63
  const int lcol = (tid & 3) * 8;   // 0,8,16,24

  for (int k0 = 0; k0 < DM; k0 += 32) {
    const float* __restrict__ xs = x + (m0 + lrow) * DM + k0 + lcol;
    const float* __restrict__ wsrc = W + (n0 + lrow) * DM + k0 + lcol;
    f32x4 xa = *(const f32x4*)xs;
    f32x4 xb = *(const f32x4*)(xs + 4);
    f32x4 wa = *(const f32x4*)wsrc;
    f32x4 wb = *(const f32x4*)(wsrc + 4);
    __syncthreads();  // previous iteration's fragment reads done
    #pragma unroll
    for (int i = 0; i < 4; ++i) {
      Xs[lrow][lcol + i]     = f32_to_bf16(xa[i]);
      Xs[lrow][lcol + 4 + i] = f32_to_bf16(xb[i]);
      Ws[lrow][lcol + i]     = f32_to_bf16(wa[i]);
      Ws[lrow][lcol + 4 + i] = f32_to_bf16(wb[i]);
    }
    __syncthreads();
    bf16x16 a  = load_frag(&Xs[mt * 16 + ln][0], half);        // A: x tile
    bf16x16 b0 = load_frag(&Ws[nt0 * 16 + ln][0], half);       // B[k][n] = W[n][k]
    bf16x16 b1 = load_frag(&Ws[(nt0 + 1) * 16 + ln][0], half);
    acc0 = wmma_bf16(a, b0, acc0);
    acc1 = wmma_bf16(a, b1, acc1);
  }

  if (which < 2) {
    u16* __restrict__ dst = (which == 0) ? Qb : Kb;
    #pragma unroll
    for (int r = 0; r < 8; ++r) {               // C layout: M = r + 8*half, N = ln
      int row = m0 + mt * 16 + 8 * half + r;
      dst[row * DM + n0 + nt0 * 16 + ln]       = f32_to_bf16(acc0[r]);
      dst[row * DM + n0 + (nt0 + 1) * 16 + ln] = f32_to_bf16(acc1[r]);
    }
  } else {
    u16x8 p0, p1;
    #pragma unroll
    for (int r = 0; r < 8; ++r) { p0[r] = f32_to_bf16(acc0[r]); p1[r] = f32_to_bf16(acc1[r]); }
    const int mbase = m0 + mt * 16 + 8 * half;  // 8 consecutive seq positions
    *(u16x8*)(Vt + (n0 + nt0 * 16 + ln) * SEQ + mbase)       = p0;
    *(u16x8*)(Vt + (n0 + (nt0 + 1) * 16 + ln) * SEQ + mbase) = p1;
  }
}

// ---------------------------------------------------------------------------
// Kernel 2: flash attention. Block = 256 threads (8 waves), 32 query rows.
// Wave w: mi = w&1 (16-query group), ci = w>>1 (192-wide output slice).
// Waves 0-3 compute the 32x32 score tile (one 16x16 each, D=768 k-loop),
// all waves do online softmax + 12 P@V WMMAs into 16x192 f32 accumulators.
// ---------------------------------------------------------------------------
__global__ __launch_bounds__(256) void flash_attn_kernel(
    const u16* __restrict__ Qb, const u16* __restrict__ Kb,
    const u16* __restrict__ Vt, float* __restrict__ out)
{
  __shared__ u16  Qs[32 * DM];    // 48 KB: query tile, bf16
  __shared__ float Sl[32][32];    // 4 KB: raw scores (scaled)

  const int tid  = threadIdx.x;
  const int lane = tid & 31;
  const int wid  = tid >> 5;
  const int half = lane >> 4;
  const int ln   = lane & 15;
  const int q0   = blockIdx.x * 32;
  const int mi   = wid & 1;
  const int ci   = wid >> 1;

  // Stage Q tile (48 KB) into LDS. Prefer CDNA5 async memory->LDS DMA
  // (no VGPR round trip, tracked by ASYNCcnt); fall back to VGPR copy.
#if HAVE_ASYNC_LDS
  {
    gas_i32x4* gsrc = (gas_i32x4*)(Qb + q0 * DM);
    las_i32x4* ldst = (las_i32x4*)Qs;
    #pragma unroll
    for (int i = 0; i < 12; ++i) {
      const int e = tid + 256 * i;       // 16-byte vector index
      __builtin_amdgcn_global_load_async_to_lds_b128(gsrc + e, ldst + e, 0, 0);
    }
  #if __has_builtin(__builtin_amdgcn_s_wait_asynccnt)
    __builtin_amdgcn_s_wait_asynccnt(0);
  #else
    asm volatile("s_wait_asynccnt 0" ::: "memory");
  #endif
  }
#else
  {
    const uint4* __restrict__ src = (const uint4*)(Qb + q0 * DM);
    uint4* dst = (uint4*)Qs;
    #pragma unroll
    for (int i = 0; i < 12; ++i) dst[tid + 256 * i] = src[tid + 256 * i];
  }
#endif
  __syncthreads();

  f32x8 O[12];
  const f32x8 zero = {};
  #pragma unroll
  for (int t = 0; t < 12; ++t) O[t] = zero;
  float m_run = -1e30f;
  float l_run = 0.0f;
  const float scale = 0.036084391824351615f;   // 1/sqrt(768)

  for (int kv0 = 0; kv0 < SEQ; kv0 += 32) {
    if (wid < 4) {                              // wave-uniform: EXEC all-ones
      const int mi2 = wid & 1;
      const int nj  = wid >> 1;
      const u16* __restrict__ qrow = Qs + (mi2 * 16 + ln) * DM;
      const u16* __restrict__ krow = Kb + (kv0 + nj * 16 + ln) * DM;
      if (kv0 + 32 < SEQ)
        __builtin_prefetch(Kb + (kv0 + 32 + nj * 16 + ln) * DM, 0, 0);
      f32x8 s = zero;
      #pragma unroll 4
      for (int k0 = 0; k0 < DM; k0 += 32) {     // B[k][n] = K[n][k] (NT gemm)
        bf16x16 a = load_frag(qrow + k0, half);
        bf16x16 b = load_frag(krow + k0, half);
        s = wmma_bf16(a, b, s);
      }
      #pragma unroll
      for (int r = 0; r < 8; ++r)
        Sl[mi2 * 16 + 8 * half + r][nj * 16 + ln] = s[r] * scale;
    }
    __syncthreads();

    // Online softmax: this lane owns row (mi*16 + ln); its 16 score cols
    // follow the A-fragment K pattern, so the exp'd values drop straight
    // into the P fragment.
    const float* __restrict__ srow = &Sl[mi * 16 + ln][0];
    f32x8 s_lo = *(const f32x8*)(srow + 8 * half);
    f32x8 s_hi = *(const f32x8*)(srow + 16 + 8 * half);
    float mx = -1e30f;
    #pragma unroll
    for (int r = 0; r < 8; ++r) { mx = fmaxf(mx, s_lo[r]); mx = fmaxf(mx, s_hi[r]); }
    mx = fmaxf(mx, __shfl_xor(mx, 16, 32));     // combine the two lane halves
    const float m_new = fmaxf(m_run, mx);
    const float alpha = __expf(m_run - m_new);
    float psum = 0.0f;
    u16x16 pv;
    #pragma unroll
    for (int r = 0; r < 8; ++r) {
      float p0 = __expf(s_lo[r] - m_new);
      float p1 = __expf(s_hi[r] - m_new);
      psum += p0 + p1;
      pv[r]     = f32_to_bf16(p0);
      pv[8 + r] = f32_to_bf16(p1);
    }
    psum += __shfl_xor(psum, 16, 32);
    m_run = m_new;
    l_run = l_run * alpha + psum;
    const bf16x16 P = __builtin_bit_cast(bf16x16, pv);

    float a8[8];                                 // alpha for accum rows M=r+8*half
    #pragma unroll
    for (int r = 0; r < 8; ++r) a8[r] = __shfl(alpha, 8 * half + r, 32);
    #pragma unroll
    for (int t = 0; t < 12; ++t) {
      #pragma unroll
      for (int r = 0; r < 8; ++r) O[t][r] *= a8[r];
    }
    if (kv0 + 32 < SEQ)                          // warm L2->L0 for next V block
      __builtin_prefetch(Vt + (ci * 192 + ln) * SEQ + kv0 + 32, 0, 0);
    #pragma unroll
    for (int t = 0; t < 12; ++t) {               // B[k][n] = Vt[col][kv0+k]
      const u16* __restrict__ vrow = Vt + (ci * 192 + t * 16 + ln) * SEQ + kv0;
      bf16x16 B = load_frag(vrow, half);
      O[t] = wmma_bf16(P, B, O[t]);
    }
    __syncthreads();                             // Sl reused next iteration
  }

  float linv[8];
  #pragma unroll
  for (int r = 0; r < 8; ++r) linv[r] = 1.0f / __shfl(l_run, 8 * half + r, 32);
  #pragma unroll
  for (int t = 0; t < 12; ++t) {
    #pragma unroll
    for (int r = 0; r < 8; ++r) {
      int row = q0 + mi * 16 + 8 * half + r;
      int col = ci * 192 + t * 16 + ln;
      out[row * DM + col] = O[t][r] * linv[r];
    }
  }
}

// ---------------------------------------------------------------------------
extern "C" void kernel_launch(void* const* d_in, const int* in_sizes, int n_in,
                              void* d_out, int out_size, void* d_ws, size_t ws_size,
                              hipStream_t stream) {
  const float* x  = (const float*)d_in[0];
  const float* wq = (const float*)d_in[1];
  const float* wk = (const float*)d_in[2];
  const float* wv = (const float*)d_in[3];

  u16* Qb = (u16*)d_ws;                          // [SEQ][DM] bf16
  u16* Kb = Qb + (size_t)SEQ * DM;               // [SEQ][DM] bf16
  u16* Vt = Kb + (size_t)SEQ * DM;               // [DM][SEQ] bf16 (transposed V)
  float* out = (float*)d_out;

  qkv_proj_kernel<<<dim3(DM / 64, SEQ / 64, 3), 256, 0, stream>>>(
      x, wq, wk, wv, Qb, Kb, Vt);
  flash_attn_kernel<<<dim3(SEQ / 32), 256, 0, stream>>>(Qb, Kb, Vt, out);
}
